// Mini_pointgnn_v15_67310727463249
// MI455X (gfx1250) — matrix operand order repaired and checked by reading, not compiled, not run on hardware
//
#include <hip/hip_runtime.h>
#include <hip/hip_bf16.h>
#include <math.h>
#include <stdint.h>

// MI455X / gfx1250, wave32. All MLPs run on v_wmma_f32_16x16x32_f16:
// 16-row tiles, K padded to 32, N=64 as 4 column tiles, weights staged in LDS
// in fragment order (2x ds_load_b128 per WMMA operand). Node features are
// mirrored in f16 in global memory so the hot edge-gather uses
// global_load_async_to_lds_b128 (ASYNCcnt) straight into the LDS staging tile.

#define WAVES 4
#define TPB   128

typedef __attribute__((ext_vector_type(16))) _Float16 v16h;
typedef __attribute__((ext_vector_type(8)))  _Float16 v8h;
typedef __attribute__((ext_vector_type(8)))  float    v8f;

static constexpr int Pn = 200000, N1n = 40000, N2n = 4000,
                     E1n = 640000, E2n = 64000, NCLSn = 20;

#define DS_FENCE()    asm volatile("s_wait_dscnt 0" ::: "memory")
#define ASYNC_FENCE() asm volatile("s_wait_asynccnt 0" ::: "memory")

// Async 16B copy: global (f16 features) -> LDS staging tile. VDST = LDS byte
// address (low 32 bits of flat shared pointer), VADDR = 64-bit global address.
__device__ __forceinline__ void async_ld16(const _Float16* lds_dst, const _Float16* g) {
  unsigned loff = (unsigned)(uintptr_t)(const void*)lds_dst;
  asm volatile("global_load_async_to_lds_b128 %0, %1, off"
               :: "v"(loff), "v"((unsigned long long)(uintptr_t)g)
               : "memory");
}

// ---- WMMA fragment builders (CDNA5 ISA 7.12.2 layouts) ----------------------
__device__ __forceinline__ v16h a_frag(const _Float16* X, int kbase, int stride, int lane) {
  const _Float16* row = X + (lane & 15) * stride + kbase + ((lane < 16) ? 0 : 8);
  v8h lo = *(const v8h*)row;
  v8h hi = *(const v8h*)(row + 16);
  return __builtin_shufflevector(lo, hi, 0, 1, 2, 3, 4, 5, 6, 7,
                                 8, 9, 10, 11, 12, 13, 14, 15);
}
__device__ __forceinline__ v16h b_frag(const _Float16* W, int tile, int lane) {
  const _Float16* p = W + ((size_t)tile * 32 + lane) * 16;
  v8h lo = *(const v8h*)p;
  v8h hi = *(const v8h*)(p + 8);
  return __builtin_shufflevector(lo, hi, 0, 1, 2, 3, 4, 5, 6, 7,
                                 8, 9, 10, 11, 12, 13, 14, 15);
}

template <int NT>
__device__ __forceinline__ void gemm16(const _Float16* X, int kpad,
                                       const _Float16* W, v8f* acc, int lane) {
  const int KT = kpad >> 5;
  for (int kt = 0; kt < KT; ++kt) {
    v16h a = a_frag(X, kt * 32, kpad, lane);
#pragma unroll
    for (int nt = 0; nt < NT; ++nt) {
      v16h b = b_frag(W, kt * NT + nt, lane);
      acc[nt] = __builtin_amdgcn_wmma_f32_16x16x32_f16(
          false, a, false, b, (short)0, acc[nt], false, false);
    }
  }
}

__device__ __forceinline__ void store_mid_f16(_Float16* Y, const v8f* acc,
                                              const float* bias, bool relu, int lane) {
  int n0 = lane & 15, mo = (lane < 16) ? 0 : 8;
#pragma unroll
  for (int nt = 0; nt < 4; ++nt) {
    int n = nt * 16 + n0;
    float bv = bias[n];
#pragma unroll
    for (int r = 0; r < 8; ++r) {
      float v = acc[nt][r] + bv;
      if (relu) v = fmaxf(v, 0.f);
      Y[(mo + r) * 64 + n] = (_Float16)v;
    }
  }
}

__device__ __forceinline__ void atomicMaxF(float* addr, float v) {
  if (v >= 0.f) atomicMax((int*)addr, __float_as_int(v));
  else          atomicMin((unsigned int*)addr, (unsigned int)__float_as_int(v));
}

// ---- small utility kernels --------------------------------------------------
__global__ void k_cvt_swz(const float* __restrict__ src, _Float16* __restrict__ dst,
                          int K, int N, int KP, int NP) {
  int i = blockIdx.x * blockDim.x + threadIdx.x;
  if (i >= KP * NP) return;
  int k = i / NP, n = i % NP;
  float v = (k < K && n < N) ? src[k * N + n] : 0.f;
  int kt = k >> 5, kk = k & 31, nt = n >> 4, nn = n & 15;
  int lane = nn + (((kk >> 3) & 1) ? 16 : 0);
  int elem = (kk & 7) + ((kk >> 4) << 3);
  int NT = NP >> 4;
  dst[(((size_t)(kt * NT + nt) * 32) + lane) * 16 + elem] = (_Float16)v;
}
__global__ void k_fill(float* p, float v, int n) {
  int i = blockIdx.x * blockDim.x + threadIdx.x;
  if (i < n) p[i] = v;
}
__global__ void k_fixinf(float* p, int n) {
  int i = blockIdx.x * blockDim.x + threadIdx.x;
  if (i < n && p[i] < -3.0e38f) p[i] = 0.f;
}
__global__ void k_fix_cvt(float* __restrict__ p, _Float16* __restrict__ ph, int n) {
  int i = blockIdx.x * blockDim.x + threadIdx.x;
  if (i >= n) return;
  float v = p[i];
  if (v < -3.0e38f) v = 0.f;
  p[i] = v;
  ph[i] = (_Float16)v;
}
__global__ void k_relm(const float* __restrict__ l1c, const float* __restrict__ l2c,
                       const int* __restrict__ lab, float* __restrict__ relm, int n) {
  int i = blockIdx.x * blockDim.x + threadIdx.x;
  if (i >= n) return;
  int j = lab[i];
#pragma unroll
  for (int d = 0; d < 3; ++d) relm[i * 3 + d] = l1c[i * 3 + d] - l2c[j * 3 + d];
}

// ---- point embedding: MLP([rel1, remission]) -> atomicAdd segment-sum -------
__global__ void __launch_bounds__(TPB) k_point_embed(
    const float* __restrict__ rem, const float* __restrict__ pts,
    const float* __restrict__ l1c, const int* __restrict__ lab,
    const _Float16* __restrict__ W1, const float* __restrict__ b1,
    const _Float16* __restrict__ W2, const float* __restrict__ b2,
    float* __restrict__ segsum, int n) {
  __shared__ __align__(16) _Float16 sW1[32 * 64];
  __shared__ __align__(16) _Float16 sW2[64 * 64];
  __shared__ __align__(16) _Float16 sX[WAVES][16 * 32];
  __shared__ __align__(16) _Float16 sM[WAVES][16 * 64];
  for (int i = threadIdx.x; i < 32 * 64; i += TPB) sW1[i] = W1[i];
  for (int i = threadIdx.x; i < 64 * 64; i += TPB) sW2[i] = W2[i];
  __syncthreads();
  int lane = threadIdx.x & 31, wave = threadIdx.x >> 5;
  int tiles = (n + 15) >> 4;
  for (int t = blockIdx.x * WAVES + wave; t < tiles; t += gridDim.x * WAVES) {
    int p0 = t << 4;
    _Float16* X = sX[wave];
    for (int idx = lane; idx < 16 * 32; idx += 32) {
      int r = idx >> 5, c = idx & 31, p = p0 + r;
      float v = 0.f;
      if (p < n) {
        if (c < 3)       v = pts[p * 3 + c] - l1c[lab[p] * 3 + c];
        else if (c == 3) v = rem[p];
      }
      X[idx] = (_Float16)v;
    }
    DS_FENCE();
    v8f acc[4] = {};
    gemm16<4>(X, 32, sW1, acc, lane);
    store_mid_f16(sM[wave], acc, b1, true, lane);
    DS_FENCE();
    v8f acc2[4] = {};
    gemm16<4>(sM[wave], 64, sW2, acc2, lane);
    int n0 = lane & 15, mo = (lane < 16) ? 0 : 8;
#pragma unroll
    for (int nt = 0; nt < 4; ++nt) {
      int col = nt * 16 + n0;
      float bv = b2[col];
#pragma unroll
      for (int r = 0; r < 8; ++r) {
        int p = p0 + mo + r;
        if (p < n) atomicAdd(&segsum[(size_t)lab[p] * 64 + col], acc2[nt][r] + bv);
      }
    }
  }
}

// ---- node MLP 64->64->64, optional residuals; writes f32 + f16 mirror ------
__global__ void __launch_bounds__(TPB) k_node_mlp(
    const float* __restrict__ in,
    const _Float16* __restrict__ W1, const float* __restrict__ b1,
    const _Float16* __restrict__ W2, const float* __restrict__ b2,
    const float* __restrict__ res1, const float* __restrict__ res2,
    float* __restrict__ out, _Float16* __restrict__ outh, int n) {
  __shared__ __align__(16) _Float16 sW1[64 * 64];
  __shared__ __align__(16) _Float16 sW2[64 * 64];
  __shared__ __align__(16) _Float16 sX[WAVES][16 * 64];
  __shared__ __align__(16) _Float16 sM[WAVES][16 * 64];
  for (int i = threadIdx.x; i < 64 * 64; i += TPB) { sW1[i] = W1[i]; sW2[i] = W2[i]; }
  __syncthreads();
  int lane = threadIdx.x & 31, wave = threadIdx.x >> 5;
  int tiles = (n + 15) >> 4;
  for (int t = blockIdx.x * WAVES + wave; t < tiles; t += gridDim.x * WAVES) {
    int r0 = t << 4;
    _Float16* X = sX[wave];
    for (int u = lane; u < 256; u += 32) {       // 16 rows x 16 float4 segs
      int row = u >> 4, seg = u & 15;
      int i = r0 + row;
      float4 val = make_float4(0.f, 0.f, 0.f, 0.f);
      if (i < n) val = *(const float4*)(in + (size_t)i * 64 + seg * 4);
      _Float16* xr = X + row * 64 + seg * 4;
      xr[0] = (_Float16)val.x; xr[1] = (_Float16)val.y;
      xr[2] = (_Float16)val.z; xr[3] = (_Float16)val.w;
    }
    DS_FENCE();
    v8f acc[4] = {};
    gemm16<4>(X, 64, sW1, acc, lane);
    store_mid_f16(sM[wave], acc, b1, true, lane);
    DS_FENCE();
    v8f acc2[4] = {};
    gemm16<4>(sM[wave], 64, sW2, acc2, lane);
    int n0 = lane & 15, mo = (lane < 16) ? 0 : 8;
#pragma unroll
    for (int nt = 0; nt < 4; ++nt) {
      int col = nt * 16 + n0;
      float bv = b2[col];
#pragma unroll
      for (int r = 0; r < 8; ++r) {
        int row = r0 + mo + r;
        if (row < n) {
          float v = acc2[nt][r] + bv;
          size_t o = (size_t)row * 64 + col;
          if (res1) v += res1[o];
          if (res2) v += res2[o];
          out[o] = v;
          if (outh) outh[o] = (_Float16)v;
        }
      }
    }
  }
}

// ---- concat MLP [Fh(gathered), rel] (67->96 pad) -> store or scatter-max ----
__global__ void __launch_bounds__(TPB) k_concat_mlp(
    const _Float16* __restrict__ Fh, const int* __restrict__ gidx,
    const float* __restrict__ rel,
    const _Float16* __restrict__ W1, const float* __restrict__ b1,
    const _Float16* __restrict__ W2, const float* __restrict__ b2,
    const int* __restrict__ sidx, float* __restrict__ out,
    _Float16* __restrict__ outh, int n) {
  __shared__ __align__(16) _Float16 sW1[96 * 64];
  __shared__ __align__(16) _Float16 sW2[64 * 64];
  __shared__ __align__(16) _Float16 sX[WAVES][16 * 96];
  __shared__ __align__(16) _Float16 sM[WAVES][16 * 64];
  for (int i = threadIdx.x; i < 96 * 64; i += TPB) sW1[i] = W1[i];
  for (int i = threadIdx.x; i < 64 * 64; i += TPB) sW2[i] = W2[i];
  int lane = threadIdx.x & 31, wave = threadIdx.x >> 5;
  for (int idx = lane; idx < 16 * 32; idx += 32)    // zero pad cols 64..95 once
    sX[wave][(idx >> 5) * 96 + 64 + (idx & 31)] = (_Float16)0.f;
  __syncthreads();
  int tiles = (n + 15) >> 4;
  for (int t = blockIdx.x * WAVES + wave; t < tiles; t += gridDim.x * WAVES) {
    int r0 = t << 4;
    _Float16* X = sX[wave];
    for (int u = lane; u < 128; u += 32) {          // 16 rows x 8 16B segs
      int row = u >> 3, seg = u & 7;
      int i = r0 + row;
      v8h val = {};
      if (i < n) {
        int fi = gidx ? gidx[i] : i;
        val = *(const v8h*)(Fh + (size_t)fi * 64 + seg * 8);
      }
      *(v8h*)(X + row * 96 + seg * 8) = val;
    }
    if (lane < 16) {
      int i = r0 + lane;
      _Float16* xr = X + lane * 96 + 64;
      if (i < n) {
        xr[0] = (_Float16)rel[i * 3 + 0];
        xr[1] = (_Float16)rel[i * 3 + 1];
        xr[2] = (_Float16)rel[i * 3 + 2];
      } else { xr[0] = xr[1] = xr[2] = (_Float16)0.f; }
    }
    DS_FENCE();
    v8f acc[4] = {};
    gemm16<4>(X, 96, sW1, acc, lane);
    store_mid_f16(sM[wave], acc, b1, true, lane);
    DS_FENCE();
    v8f acc2[4] = {};
    gemm16<4>(sM[wave], 64, sW2, acc2, lane);
    int n0 = lane & 15, mo = (lane < 16) ? 0 : 8;
#pragma unroll
    for (int nt = 0; nt < 4; ++nt) {
      int col = nt * 16 + n0;
      float bv = b2[col];
#pragma unroll
      for (int r = 0; r < 8; ++r) {
        int row = r0 + mo + r;
        if (row < n) {
          float v = acc2[nt][r] + bv;
          if (sidx) {
            atomicMaxF(&out[(size_t)sidx[row] * 64 + col], v);
          } else {
            size_t o = (size_t)row * 64 + col;
            out[o] = v;
            if (outh) outh[o] = (_Float16)v;
          }
        }
      }
    }
  }
}

// ---- GNN edge MLP: [fh[s], c[s]-c[d]] -> 64 -> atomicMax agg[d] -------------
// Source features arrive via global_load_async_to_lds_b128 (ASYNCcnt).
__global__ void __launch_bounds__(TPB) k_gnn_edge(
    const _Float16* __restrict__ Fh, const float* __restrict__ centers,
    const int* __restrict__ edges,
    const _Float16* __restrict__ W1, const float* __restrict__ b1,
    const _Float16* __restrict__ W2, const float* __restrict__ b2,
    float* __restrict__ agg, int E) {
  __shared__ __align__(16) _Float16 sW1[96 * 64];
  __shared__ __align__(16) _Float16 sW2[64 * 64];
  __shared__ __align__(16) _Float16 sX[WAVES][16 * 96];
  __shared__ __align__(16) _Float16 sM[WAVES][16 * 64];
  for (int i = threadIdx.x; i < 96 * 64; i += TPB) sW1[i] = W1[i];
  for (int i = threadIdx.x; i < 64 * 64; i += TPB) sW2[i] = W2[i];
  int lane = threadIdx.x & 31, wave = threadIdx.x >> 5;
  for (int idx = lane; idx < 16 * 32; idx += 32)    // zero pad cols 64..95 once
    sX[wave][(idx >> 5) * 96 + 64 + (idx & 31)] = (_Float16)0.f;
  __syncthreads();
  int tiles = (E + 15) >> 4;
  for (int t = blockIdx.x * WAVES + wave; t < tiles; t += gridDim.x * WAVES) {
    int e0 = t << 4;
    _Float16* X = sX[wave];
    // 16 rows x 128B features = 128 x 16B segments; 4 async instructions.
#pragma unroll
    for (int i = 0; i < 4; ++i) {
      int u = i * 32 + lane;
      int row = u >> 3, seg = u & 7;
      int e = e0 + row;
      int s = (e < E) ? edges[2 * e] : 0;
      async_ld16(X + row * 96 + seg * 8, Fh + (size_t)s * 64 + seg * 8);
    }
    if (lane < 16) {
      int e = e0 + lane;
      _Float16* xr = X + lane * 96 + 64;
      if (e < E) {
        int s = edges[2 * e], d = edges[2 * e + 1];
        xr[0] = (_Float16)(centers[s * 3 + 0] - centers[d * 3 + 0]);
        xr[1] = (_Float16)(centers[s * 3 + 1] - centers[d * 3 + 1]);
        xr[2] = (_Float16)(centers[s * 3 + 2] - centers[d * 3 + 2]);
      } else { xr[0] = xr[1] = xr[2] = (_Float16)0.f; }
    }
    ASYNC_FENCE();
    DS_FENCE();
    v8f acc[4] = {};
    gemm16<4>(X, 96, sW1, acc, lane);
    store_mid_f16(sM[wave], acc, b1, true, lane);
    DS_FENCE();
    v8f acc2[4] = {};
    gemm16<4>(sM[wave], 64, sW2, acc2, lane);
    int n0 = lane & 15, mo = (lane < 16) ? 0 : 8;
#pragma unroll
    for (int nt = 0; nt < 4; ++nt) {
      int col = nt * 16 + n0;
      float bv = b2[col];
#pragma unroll
      for (int r = 0; r < 8; ++r) {
        int e = e0 + mo + r;
        if (e < E) {
          int d = edges[2 * e + 1];
          atomicMaxF(&agg[(size_t)d * 64 + col], acc2[nt][r] + bv);
        }
      }
    }
  }
}

// ---- fused point head: fbn (67->64->64) then cls (64->64->20) ---------------
__global__ void __launch_bounds__(TPB) k_final(
    const _Float16* __restrict__ t6h, const int* __restrict__ lab,
    const float* __restrict__ pts, const float* __restrict__ l1c,
    const _Float16* __restrict__ fW1, const float* __restrict__ fb1,
    const _Float16* __restrict__ fW2, const float* __restrict__ fb2,
    const _Float16* __restrict__ cW1, const float* __restrict__ cb1,
    const _Float16* __restrict__ cW2, const float* __restrict__ cb2,
    float* __restrict__ out, int n) {
  __shared__ __align__(16) _Float16 s1[96 * 64], s2[64 * 64], s3[64 * 64], s4[64 * 32];
  __shared__ __align__(16) _Float16 sX[WAVES][16 * 96];
  __shared__ __align__(16) _Float16 sA[WAVES][16 * 64];
  __shared__ __align__(16) _Float16 sB[WAVES][16 * 64];
  for (int i = threadIdx.x; i < 96 * 64; i += TPB) s1[i] = fW1[i];
  for (int i = threadIdx.x; i < 64 * 64; i += TPB) { s2[i] = fW2[i]; s3[i] = cW1[i]; }
  for (int i = threadIdx.x; i < 64 * 32; i += TPB) s4[i] = cW2[i];
  int lane = threadIdx.x & 31, wave = threadIdx.x >> 5;
  for (int idx = lane; idx < 16 * 32; idx += 32)    // zero pad cols 64..95 once
    sX[wave][(idx >> 5) * 96 + 64 + (idx & 31)] = (_Float16)0.f;
  __syncthreads();
  int tiles = (n + 15) >> 4;
  for (int t = blockIdx.x * WAVES + wave; t < tiles; t += gridDim.x * WAVES) {
    int p0 = t << 4;
    _Float16* X = sX[wave];
    for (int u = lane; u < 128; u += 32) {
      int row = u >> 3, seg = u & 7;
      int p = p0 + row;
      v8h val = {};
      if (p < n) val = *(const v8h*)(t6h + (size_t)lab[p] * 64 + seg * 8);
      *(v8h*)(X + row * 96 + seg * 8) = val;
    }
    if (lane < 16) {
      int p = p0 + lane;
      _Float16* xr = X + lane * 96 + 64;
      if (p < n) {
        int l = lab[p];
        xr[0] = (_Float16)(pts[p * 3 + 0] - l1c[l * 3 + 0]);
        xr[1] = (_Float16)(pts[p * 3 + 1] - l1c[l * 3 + 1]);
        xr[2] = (_Float16)(pts[p * 3 + 2] - l1c[l * 3 + 2]);
      } else { xr[0] = xr[1] = xr[2] = (_Float16)0.f; }
    }
    DS_FENCE();
    v8f a1[4] = {};
    gemm16<4>(X, 96, s1, a1, lane);
    store_mid_f16(sA[wave], a1, fb1, true, lane);
    DS_FENCE();
    v8f a2[4] = {};
    gemm16<4>(sA[wave], 64, s2, a2, lane);
    store_mid_f16(sB[wave], a2, fb2, false, lane);
    DS_FENCE();
    v8f a3[4] = {};
    gemm16<4>(sB[wave], 64, s3, a3, lane);
    store_mid_f16(sA[wave], a3, cb1, true, lane);
    DS_FENCE();
    v8f a4[2] = {};
    gemm16<2>(sA[wave], 64, s4, a4, lane);
    int n0 = lane & 15, mo = (lane < 16) ? 0 : 8;
#pragma unroll
    for (int nt = 0; nt < 2; ++nt) {
      int col = nt * 16 + n0;
#pragma unroll
      for (int r = 0; r < 8; ++r) {
        int p = p0 + mo + r;
        if (p < n && col < NCLSn) out[(size_t)p * NCLSn + col] = a4[nt][r] + cb2[col];
      }
    }
  }
}

// ---- host orchestration -----------------------------------------------------
extern "C" void kernel_launch(void* const* d_in, const int* in_sizes, int n_in,
                              void* d_out, int out_size, void* d_ws, size_t ws_size,
                              hipStream_t stream) {
  (void)in_sizes; (void)n_in; (void)out_size; (void)ws_size;
  const float* rem    = (const float*)d_in[0];
  const float* pts    = (const float*)d_in[1];
  const float* l1c    = (const float*)d_in[2];
  const float* l2c    = (const float*)d_in[3];
  const int*   l1e    = (const int*)d_in[4];
  const int*   l2e    = (const int*)d_in[5];
  const int*   l1lab  = (const int*)d_in[6];
  const int*   l2lab  = (const int*)d_in[7];
  const float* ffn_eW1 = (const float*)d_in[8];   const float* ffn_eb1 = (const float*)d_in[9];
  const float* ffn_eW2 = (const float*)d_in[10];  const float* ffn_eb2 = (const float*)d_in[11];
  const float* ffn_oW1 = (const float*)d_in[12];  const float* ffn_ob1 = (const float*)d_in[13];
  const float* ffn_oW2 = (const float*)d_in[14];  const float* ffn_ob2 = (const float*)d_in[15];
  const float* g_eW1 = (const float*)d_in[16];    const float* g_eb1 = (const float*)d_in[17];
  const float* g_eW2 = (const float*)d_in[18];    const float* g_eb2 = (const float*)d_in[19];
  const float* g_oW1 = (const float*)d_in[20];    const float* g_ob1 = (const float*)d_in[21];
  const float* g_oW2 = (const float*)d_in[22];    const float* g_ob2 = (const float*)d_in[23];
  const float* m2l_W1 = (const float*)d_in[24];   const float* m2l_b1 = (const float*)d_in[25];
  const float* m2l_W2 = (const float*)d_in[26];   const float* m2l_b2 = (const float*)d_in[27];
  const float* l2m_W1 = (const float*)d_in[28];   const float* l2m_b1 = (const float*)d_in[29];
  const float* l2m_W2 = (const float*)d_in[30];   const float* l2m_b2 = (const float*)d_in[31];
  const float* fbn_W1 = (const float*)d_in[32];   const float* fbn_b1 = (const float*)d_in[33];
  const float* fbn_W2 = (const float*)d_in[34];   const float* fbn_b2 = (const float*)d_in[35];
  const float* cls_W1 = (const float*)d_in[36];   const float* cls_b1 = (const float*)d_in[37];
  const float* cls_W2 = (const float*)d_in[38];   const float* cls_b2 = (const float*)d_in[39];

  char* ws = (char*)d_ws;
  size_t off = 0;
  auto alloc  = [&](size_t bytes) -> void* { off = (off + 255) & ~(size_t)255;
                                             void* p = ws + off; off += bytes; return p; };
  auto halloc = [&](size_t e) -> _Float16* { return (_Float16*)alloc(e * sizeof(_Float16)); };
  auto falloc = [&](size_t e) -> float*    { return (float*)alloc(e * sizeof(float)); };
  auto cvt = [&](const float* src, _Float16* dst, int K, int N, int KP, int NP) {
    int total = KP * NP;
    k_cvt_swz<<<(total + 255) / 256, 256, 0, stream>>>(src, dst, K, N, KP, NP);
  };
  auto fill = [&](float* p, float v, int ne) {
    k_fill<<<(ne + 255) / 256, 256, 0, stream>>>(p, v, ne);
  };
  auto fixinf = [&](float* p, int ne) {
    k_fixinf<<<(ne + 255) / 256, 256, 0, stream>>>(p, ne);
  };
  auto gb = [&](int rows) { int t = (rows + 15) / 16; int b = (t + WAVES - 1) / WAVES;
                            if (b < 1) b = 1; if (b > 4096) b = 4096; return b; };

  // f16 weights (K padded to 32-multiple, N padded to 16-multiple, swizzled)
  _Float16* ffe1h = halloc(32 * 64);  cvt(ffn_eW1, ffe1h, 4, 64, 32, 64);
  _Float16* ffe2h = halloc(64 * 64);  cvt(ffn_eW2, ffe2h, 64, 64, 64, 64);
  _Float16* ffo1h = halloc(64 * 64);  cvt(ffn_oW1, ffo1h, 64, 64, 64, 64);
  _Float16* ffo2h = halloc(64 * 64);  cvt(ffn_oW2, ffo2h, 64, 64, 64, 64);
  _Float16 *gE1h[9], *gE2h[9], *gO1h[9], *gO2h[9];
  for (int i = 0; i < 9; ++i) {
    gE1h[i] = halloc(96 * 64); cvt(g_eW1 + (size_t)i * 67 * 64, gE1h[i], 67, 64, 96, 64);
    gE2h[i] = halloc(64 * 64); cvt(g_eW2 + (size_t)i * 64 * 64, gE2h[i], 64, 64, 64, 64);
    gO1h[i] = halloc(64 * 64); cvt(g_oW1 + (size_t)i * 64 * 64, gO1h[i], 64, 64, 64, 64);
    gO2h[i] = halloc(64 * 64); cvt(g_oW2 + (size_t)i * 64 * 64, gO2h[i], 64, 64, 64, 64);
  }
  _Float16* m2l1h = halloc(96 * 64);  cvt(m2l_W1, m2l1h, 67, 64, 96, 64);
  _Float16* m2l2h = halloc(64 * 64);  cvt(m2l_W2, m2l2h, 64, 64, 64, 64);
  _Float16* l2m1h = halloc(96 * 64);  cvt(l2m_W1, l2m1h, 67, 64, 96, 64);
  _Float16* l2m2h = halloc(64 * 64);  cvt(l2m_W2, l2m2h, 64, 64, 64, 64);
  _Float16* fbn1h = halloc(96 * 64);  cvt(fbn_W1, fbn1h, 67, 64, 96, 64);
  _Float16* fbn2h = halloc(64 * 64);  cvt(fbn_W2, fbn2h, 64, 64, 64, 64);
  _Float16* cls1h = halloc(64 * 64);  cvt(cls_W1, cls1h, 64, 64, 64, 64);
  _Float16* cls2h = halloc(64 * 32);  cvt(cls_W2, cls2h, 64, 20, 64, 32);

  // f32 activation buffers + f16 feature mirrors (edge-gather operands)
  float* segsum = falloc((size_t)N1n * 64);
  float* t1  = falloc((size_t)N1n * 64);   _Float16* t1h  = halloc((size_t)N1n * 64);
  float* t2  = falloc((size_t)N1n * 64);   _Float16* t2h  = halloc((size_t)N1n * 64);
  float* t21 = falloc((size_t)N1n * 64);   _Float16* t21h = halloc((size_t)N1n * 64);
  float* t22 = falloc((size_t)N1n * 64);   _Float16* t22h = halloc((size_t)N1n * 64);
  float* t23 = falloc((size_t)N1n * 64);   _Float16* t23h = halloc((size_t)N1n * 64);
  float* cur = falloc((size_t)N1n * 64);   _Float16* curh = halloc((size_t)N1n * 64);
  float* agg = falloc((size_t)N1n * 64);
  float* relm = falloc((size_t)N1n * 3);
  float* f3 = falloc((size_t)N2n * 64);    _Float16* f3h = halloc((size_t)N2n * 64);
  float* f4 = falloc((size_t)N2n * 64);    _Float16* f4h = halloc((size_t)N2n * 64);

  const float NEG = -__builtin_huge_valf();
  auto node = [&](const float* in, const _Float16* W1, const float* b1,
                  const _Float16* W2, const float* b2,
                  const float* r1, const float* r2,
                  float* out, _Float16* outh, int nrows) {
    k_node_mlp<<<gb(nrows), TPB, 0, stream>>>(in, W1, b1, W2, b2, r1, r2, out, outh, nrows);
  };
  auto gnn_l1 = [&](const float* fin, const _Float16* finh, int li, const float* r2,
                    float* fout, _Float16* fouth) {
    fill(agg, NEG, N1n * 64);
    k_gnn_edge<<<gb(E1n), TPB, 0, stream>>>(finh, l1c, l1e, gE1h[li], g_eb1 + li * 64,
                                            gE2h[li], g_eb2 + li * 64, agg, E1n);
    fixinf(agg, N1n * 64);
    node(agg, gO1h[li], g_ob1 + li * 64, gO2h[li], g_ob2 + li * 64, fin, r2, fout, fouth, N1n);
  };

  // 1) point embed -> segment sum -> t1
  fill(segsum, 0.f, N1n * 64);
  k_point_embed<<<gb(Pn), TPB, 0, stream>>>(rem, pts, l1c, l1lab,
                                            ffe1h, ffn_eb1, ffe2h, ffn_eb2, segsum, Pn);
  node(segsum, ffo1h, ffn_ob1, ffo2h, ffn_ob2, nullptr, nullptr, t1, t1h, N1n);
  k_relm<<<(N1n + 255) / 256, 256, 0, stream>>>(l1c, l2c, l2lab, relm, N1n);

  // 2) down GNN stack on L1 graph
  gnn_l1(t1,  t1h,  0, nullptr, t2,  t2h);
  gnn_l1(t2,  t2h,  1, nullptr, t21, t21h);
  gnn_l1(t21, t21h, 2, nullptr, t22, t22h);
  gnn_l1(t22, t22h, 3, nullptr, t23, t23h);

  // 3) L1 -> L2 pooling + L2 GNN + L2 -> L1
  fill(f3, NEG, N2n * 64);
  k_concat_mlp<<<gb(N1n), TPB, 0, stream>>>(t23h, nullptr, relm, m2l1h, m2l_b1,
                                            m2l2h, m2l_b2, l2lab, f3, nullptr, N1n);
  k_fix_cvt<<<(N2n * 64 + 255) / 256, 256, 0, stream>>>(f3, f3h, N2n * 64);
  fill(agg, NEG, N2n * 64);
  k_gnn_edge<<<gb(E2n), TPB, 0, stream>>>(f3h, l2c, l2e, gE1h[4], g_eb1 + 4 * 64,
                                          gE2h[4], g_eb2 + 4 * 64, agg, E2n);
  fixinf(agg, N2n * 64);
  node(agg, gO1h[4], g_ob1 + 4 * 64, gO2h[4], g_ob2 + 4 * 64, f3, nullptr, f4, f4h, N2n);
  k_concat_mlp<<<gb(N1n), TPB, 0, stream>>>(f4h, l2lab, relm, l2m1h, l2m_b1,
                                            l2m2h, l2m_b2, nullptr, cur, curh, N1n);

  // 4) up GNN stack with skip residuals (in-place on cur/curh)
  gnn_l1(cur, curh, 5, t23, cur, curh);
  gnn_l1(cur, curh, 6, t22, cur, curh);
  gnn_l1(cur, curh, 7, t21, cur, curh);
  gnn_l1(cur, curh, 8, t2,  cur, curh);

  // 5) fused per-point head -> logits
  k_final<<<gb(Pn), TPB, 0, stream>>>(curh, l1lab, pts, l1c, fbn1h, fbn_b1, fbn2h, fbn_b2,
                                      cls1h, cls_b1, cls2h, cls_b2, (float*)d_out, Pn);
}
